// HybridA_20246475833375
// MI455X (gfx1250) — matrix operand order repaired and checked by reading, not compiled, not run on hardware
//
#include <hip/hip_runtime.h>
#include <math.h>
#include <stdint.h>

// Problem dims (fixed by reference)
#define BB 2
#define LL 1024
#define DDIM 1024
#define VV 32000
#define EDIM 2048
#define NSTATE 16
#define DCONVK 4
#define DTRANK 64
#define MM (BB * LL)  // 2048 tokens

typedef __attribute__((ext_vector_type(16))) _Float16 v16h;
typedef __attribute__((ext_vector_type(2)))  _Float16 h2;
typedef __attribute__((ext_vector_type(8)))  float    v8f;
typedef __attribute__((ext_vector_type(4)))  float    f4;

__device__ __forceinline__ float silu_f(float x)     { return x / (1.f + __expf(-x)); }
__device__ __forceinline__ float softplus_f(float x) { return x > 20.f ? x : log1pf(__expf(x)); }

// ---------------------------------------------------------------------------
// Embedding gather: x0[m, :] = embed_w[tokens[m], :]
// ---------------------------------------------------------------------------
__global__ __launch_bounds__(256) void embed_kernel(const int* __restrict__ tok,
                                                    const float* __restrict__ ew,
                                                    float* __restrict__ out) {
  int m = blockIdx.x;
  int t = threadIdx.x;
  int token = tok[m];
  const f4* src = (const f4*)(ew + (long)token * DDIM);
  f4* dst = (f4*)(out + (long)m * DDIM);
  dst[t] = src[t];  // 256 threads x float4 = 1024 floats
}

// ---------------------------------------------------------------------------
// Generic WMMA GEMM:  C = act( scale * (A @ B) + bias )
//   A:[M,K] f32 row-major (lda), B:[K,N] (ldb) or transposed [N,K] when BT.
//   64x128 C tile per block, 4 waves (128 threads), K stepped by 32.
//   f32 -> f16 conversion through LDS; B is staged K-major (sB[n][k]) so both
//   A and B fragments are two contiguous 8-half runs -> ds_load_b128 pairs.
// ---------------------------------------------------------------------------
#define BM 64
#define BN 128
#define BK 32
#define PADH 8  // halves of row padding (keeps rows 4B-aligned, breaks bank stride)

template <bool BT>
__global__ __launch_bounds__(128) void gemm_wmma(
    const float* __restrict__ A, const float* __restrict__ B,
    const float* __restrict__ bias, float* __restrict__ C,
    int Mtot, int Ntot, int Ktot, int lda, int ldb, int ldc,
    long sAb, long sBb, long sCb, float scale, int act) {
  (void)Mtot;
  __shared__ _Float16 sA[BM][BK + PADH];  // [m][k]
  __shared__ _Float16 sB[BN][BK + PADH];  // [n][k]  (K-major!)

  const int tid  = threadIdx.x;
  const int lane = tid & 31;
  const int wave = tid >> 5;  // 0..3 -> 16-row strip of the 64-row tile
  const int n0 = blockIdx.x * BN;
  const int m0 = blockIdx.y * BM;
  const int bz = blockIdx.z;

  const float* Ab = A + (long)bz * sAb;
  const float* Bb = B + (long)bz * sBb;
  float*       Cb = C + (long)bz * sCb;

  v8f acc[8];
#pragma unroll
  for (int t = 0; t < 8; ++t)
#pragma unroll
    for (int i = 0; i < 8; ++i) acc[t][i] = 0.f;

  // A staging assignment
  const int ar = tid >> 1;         // 0..63  row in tile
  const int ac = (tid & 1) * 16;   // 0/16   col seg

  for (int k0 = 0; k0 < Ktot; k0 += BK) {
    // ---- stage A tile (64x32) as f16, row-major ----
    {
      const float* src = Ab + (long)(m0 + ar) * lda + k0 + ac;
#pragma unroll
      for (int v = 0; v < 4; ++v) {
        f4 x = *(const f4*)(src + v * 4);
#pragma unroll
        for (int i = 0; i < 4; ++i) sA[ar][ac + v * 4 + i] = (_Float16)x[i];
      }
    }
    // ---- stage B tile as f16 into sB[n][k] ----
    if (!BT) {
      // B is [K,N]: read two K-adjacent rows, pack pairs -> 32-bit LDS stores.
      // 512 slots: kp (16) x n4 (32); 4 slots per thread.
#pragma unroll
      for (int i = 0; i < 4; ++i) {
        int slot = tid + 128 * i;
        int kp = slot >> 5;          // k pair index 0..15
        int n4 = (slot & 31) * 4;    // n start within tile
        int gn = n0 + n4;
        f4 r0 = {0.f, 0.f, 0.f, 0.f}, r1 = {0.f, 0.f, 0.f, 0.f};
        if (gn < Ntot) {
          r0 = *(const f4*)(Bb + (long)(k0 + 2 * kp) * ldb + gn);
          r1 = *(const f4*)(Bb + (long)(k0 + 2 * kp + 1) * ldb + gn);
        }
#pragma unroll
        for (int c = 0; c < 4; ++c) {
          h2 p = {(_Float16)r0[c], (_Float16)r1[c]};
          *(h2*)&sB[n4 + c][2 * kp] = p;
        }
      }
    } else {
      // B' is [N,K]: K-contiguous in memory -> contiguous LDS row stores.
      int gn = n0 + tid;  // 128 rows, one per thread
      const float* src = Bb + (long)gn * ldb + k0;
#pragma unroll
      for (int v = 0; v < 8; ++v) {
        f4 x = {0.f, 0.f, 0.f, 0.f};
        if (gn < Ntot) x = *(const f4*)(src + 4 * v);
        h2 p0 = {(_Float16)x[0], (_Float16)x[1]};
        h2 p1 = {(_Float16)x[2], (_Float16)x[3]};
        *(h2*)&sB[tid][4 * v]     = p0;
        *(h2*)&sB[tid][4 * v + 2] = p1;
      }
    }
    // prefetch next K tiles into cache while we compute
    if (k0 + BK < Ktot) {
      __builtin_prefetch(Ab + (long)(m0 + ar) * lda + (k0 + BK) + ac, 0, 1);
      if (!BT)
        __builtin_prefetch(Bb + (long)(k0 + BK + (tid >> 5) * 2) * ldb + n0 + (tid & 31) * 4, 0, 1);
      else
        __builtin_prefetch(Bb + (long)(n0 + tid) * ldb + (k0 + BK), 0, 1);
    }
    __syncthreads();

    // ---- build fragments & issue WMMA ----
    // A 16x32 f16 layout (ISA 7.12.2): lanes 0-15 (rows M=0..15) hold K 0..7
    // in elems 0..7 and K 16..23 in elems 8..15; lanes 16-31 hold K 8..15 /
// 24..31.  B 32x16: same K split as A (lanes 0-15 K 0..7 / 16..23;
    // lanes 16-31 K 8..15 / 24..31).
    const int r   = lane & 15;
    const int khi = lane >> 4;
    v16h af;
    {
      const _Float16* ap = &sA[wave * 16 + r][0];
#pragma unroll
      for (int j = 0; j < 8; ++j) af[j] = ap[khi * 8 + j];
#pragma unroll
      for (int j = 0; j < 8; ++j) af[8 + j] = ap[16 + khi * 8 + j];
    }
#pragma unroll
    for (int t = 0; t < 8; ++t) {
      v16h bf;
      const _Float16* bp = &sB[t * 16 + r][khi * 8];
#pragma unroll
      for (int j = 0; j < 8; ++j) bf[j] = bp[j];
#pragma unroll
      for (int j = 0; j < 8; ++j) bf[8 + j] = bp[16 + j];
      acc[t] = __builtin_amdgcn_wmma_f32_16x16x32_f16(
          false, af, false, bf, (short)0, acc[t], false, false);
    }
    __syncthreads();
  }

  // ---- epilogue: C 16x16 f32 layout: lane<16 -> M=vgpr, N=lane;
  //      lane>=16 -> M=8+vgpr, N=lane-16 ----
  const int r  = lane & 15;
  const int mh = (lane >> 4) * 8;
#pragma unroll
  for (int t = 0; t < 8; ++t) {
    int n = n0 + t * 16 + r;
    if (n < Ntot) {
      float bv = bias ? bias[n] : 0.f;
#pragma unroll
      for (int rr = 0; rr < 8; ++rr) {
        int m = m0 + wave * 16 + mh + rr;
        float v = acc[t][rr] * scale + bv;
        if (act == 1) v = softplus_f(v);
        Cb[(long)m * ldc + n] = v;
      }
    }
  }
}

// ---------------------------------------------------------------------------
// Depthwise causal conv1d (DCONV=4) + bias + SiLU.  xm = xz[:, :ED]
// ---------------------------------------------------------------------------
__global__ __launch_bounds__(256) void conv_silu_kernel(
    const float* __restrict__ xz, const float* __restrict__ cw,
    const float* __restrict__ cb, float* __restrict__ xc) {
  long i = (long)blockIdx.x * 256 + threadIdx.x;  // over MM*ED
  if (i >= (long)MM * EDIM) return;
  int e = (int)(i % EDIM);
  int m = (int)(i / EDIM);
  int b = m / LL, l = m % LL;
  float s = cb[e];
#pragma unroll
  for (int j = 0; j < DCONVK; ++j) {
    int li = l - (DCONVK - 1) + j;
    if (li >= 0) s += cw[e * DCONVK + j] * xz[(long)(b * LL + li) * (2 * EDIM) + e];
  }
  xc[i] = silu_f(s);
}

// ---------------------------------------------------------------------------
// Selective scan: one thread per (b, e) channel, h[16] in registers.
// Per-timestep B/C coefficients (32 floats) are DMA'd into LDS via the
// CDNA5 async-to-LDS path (ASYNCcnt) by wave 0.
// y = (sum_n h_n*C_n + xm*D) * silu(z)
// ---------------------------------------------------------------------------
__global__ __launch_bounds__(256) void scan_kernel(
    const float* __restrict__ dbl, const float* __restrict__ dlt,
    const float* __restrict__ xc, const float* __restrict__ xz,
    const float* __restrict__ A_log, const float* __restrict__ Dp,
    float* __restrict__ y) {
  int e = blockIdx.x * 256 + threadIdx.x;
  int b = blockIdx.y;
  __shared__ float sBC[2 * NSTATE];
  float an[NSTATE], h[NSTATE];
#pragma unroll
  for (int n = 0; n < NSTATE; ++n) {
    an[n] = -__expf(A_log[e * NSTATE + n]);
    h[n] = 0.f;
  }
  float dpe = Dp[e];
  for (int l = 0; l < LL; ++l) {
    long row = (long)b * LL + l;
    if (threadIdx.x < 2 * NSTATE) {
      unsigned lds_off = (unsigned)(uintptr_t)(&sBC[threadIdx.x]);
      const float* gp = dbl + row * (DTRANK + 2 * NSTATE) + DTRANK + threadIdx.x;
      asm volatile("global_load_async_to_lds_b32 %0, %1, off"
                   :: "v"(lds_off), "v"((unsigned long long)(uintptr_t)gp)
                   : "memory");
    }
    asm volatile("s_wait_asynccnt 0" ::: "memory");
    __syncthreads();
    float d  = dlt[row * EDIM + e];
    float xv = xc[row * EDIM + e];
    float dx = d * xv;
    float acc = 0.f;
#pragma unroll
    for (int n = 0; n < NSTATE; ++n) {
      h[n] = __expf(d * an[n]) * h[n] + dx * sBC[n];
      acc += h[n] * sBC[NSTATE + n];
    }
    float zv = xz[row * (2 * EDIM) + EDIM + e];
    y[row * EDIM + e] = (acc + xv * dpe) * silu_f(zv);
    __syncthreads();
  }
}

// ---------------------------------------------------------------------------
// LayerNorm over D=1024, one block per row
// ---------------------------------------------------------------------------
__global__ __launch_bounds__(256) void layernorm_kernel(
    const float* __restrict__ x, const float* __restrict__ w,
    const float* __restrict__ b, float* __restrict__ o) {
  int row = blockIdx.x;
  const float* xr = x + (long)row * DDIM;
  float v[4], s = 0.f, s2 = 0.f;
#pragma unroll
  for (int i = 0; i < 4; ++i) {
    v[i] = xr[threadIdx.x + i * 256];
    s += v[i];
    s2 += v[i] * v[i];
  }
  __shared__ float r1[256], r2[256];
  r1[threadIdx.x] = s;
  r2[threadIdx.x] = s2;
  __syncthreads();
  for (int st = 128; st > 0; st >>= 1) {
    if (threadIdx.x < st) {
      r1[threadIdx.x] += r1[threadIdx.x + st];
      r2[threadIdx.x] += r2[threadIdx.x + st];
    }
    __syncthreads();
  }
  float mu = r1[0] * (1.f / DDIM);
  float var = r2[0] * (1.f / DDIM) - mu * mu;
  float rstd = rsqrtf(var + 1e-5f);
  float* orow = o + (long)row * DDIM;
#pragma unroll
  for (int i = 0; i < 4; ++i) {
    int d = threadIdx.x + i * 256;
    orow[d] = (v[i] - mu) * rstd * w[d] + b[d];
  }
}

// ---------------------------------------------------------------------------
// Causal mask + softmax over a score row of length L (in place)
// ---------------------------------------------------------------------------
__global__ __launch_bounds__(256) void softmax_causal_kernel(float* __restrict__ sc) {
  int r = blockIdx.x, b = blockIdx.y, t = threadIdx.x;
  float* row = sc + ((long)b * LL + r) * LL;
  float v[4], mx = -3.4e38f;
#pragma unroll
  for (int i = 0; i < 4; ++i) {
    int c = t + i * 256;
    v[i] = (c <= r) ? row[c] : -3.4e38f;
    mx = fmaxf(mx, v[i]);
  }
  __shared__ float red[256];
  red[t] = mx;
  __syncthreads();
  for (int st = 128; st > 0; st >>= 1) {
    if (t < st) red[t] = fmaxf(red[t], red[t + st]);
    __syncthreads();
  }
  mx = red[0];
  __syncthreads();
  float s = 0.f;
#pragma unroll
  for (int i = 0; i < 4; ++i) {
    int c = t + i * 256;
    v[i] = (c <= r) ? __expf(v[i] - mx) : 0.f;
    s += v[i];
  }
  red[t] = s;
  __syncthreads();
  for (int st = 128; st > 0; st >>= 1) {
    if (t < st) red[t] += red[t + st];
    __syncthreads();
  }
  float inv = 1.f / red[0];
#pragma unroll
  for (int i = 0; i < 4; ++i) row[t + i * 256] = v[i] * inv;
}

// ---------------------------------------------------------------------------
extern "C" void kernel_launch(void* const* d_in, const int* in_sizes, int n_in,
                              void* d_out, int out_size, void* d_ws, size_t ws_size,
                              hipStream_t stream) {
  (void)in_sizes; (void)n_in; (void)out_size; (void)ws_size;
  const int*   tokens     = (const int*)d_in[0];
  const float* embed_w    = (const float*)d_in[1];
  const float* in_proj_w  = (const float*)d_in[2];
  const float* conv_w     = (const float*)d_in[3];
  const float* conv_b     = (const float*)d_in[4];
  const float* x_proj_w   = (const float*)d_in[5];
  const float* dt_proj_w  = (const float*)d_in[6];
  const float* dt_proj_b  = (const float*)d_in[7];
  const float* A_log      = (const float*)d_in[8];
  const float* D_p        = (const float*)d_in[9];
  const float* out_proj_w = (const float*)d_in[10];
  const float* ln0_w = (const float*)d_in[11];
  const float* ln0_b = (const float*)d_in[12];
  const float* ln1_w = (const float*)d_in[13];
  const float* ln1_b = (const float*)d_in[14];
  const float* c_attn_w = (const float*)d_in[15];
  const float* c_attn_b = (const float*)d_in[16];
  const float* head_w   = (const float*)d_in[17];
  const float* head_b   = (const float*)d_in[18];

  // workspace layout (floats); peak ~93 MB with aliasing
  float* ws  = (float*)d_ws;
  float* x0  = ws;                              // [2048,1024]
  float* xz  = x0 + (long)MM * DDIM;            // [2048,4096]
  float* xc  = xz + (long)MM * 2 * EDIM;        // [2048,2048]
  float* dbl = xc + (long)MM * EDIM;            // [2048,96]
  float* dlt = dbl + (long)MM * (DTRANK + 2 * NSTATE);  // [2048,2048]
  float* yv  = dlt + (long)MM * EDIM;           // [2048,2048]
  // aliases (lifetimes do not overlap)
  float* m_out = x0;   // mamba out [2048,1024]
  float* ln0o  = yv;   // yv free after out_proj
  float* qkvv  = xz;   // xz free after scan
  float* sc    = xc;   // xc free after scan
  float* att   = dlt;  // dlt free after scan
  float* ln1o  = x0;   // x0 free after ln0

  embed_kernel<<<MM, 256, 0, stream>>>(tokens, embed_w, x0);

  // in_proj: [2048,1024] @ [1024,4096] -> xz
  gemm_wmma<false><<<dim3(2 * EDIM / BN, MM / BM, 1), 128, 0, stream>>>(
      x0, in_proj_w, nullptr, xz, MM, 2 * EDIM, DDIM, DDIM, 2 * EDIM, 2 * EDIM,
      0, 0, 0, 1.f, 0);

  long tot = (long)MM * EDIM;
  conv_silu_kernel<<<(unsigned)((tot + 255) / 256), 256, 0, stream>>>(xz, conv_w, conv_b, xc);

  // x_proj: [2048,2048] @ [2048,96] -> dbl  (N=96 -> one 128-wide tile, guarded)
  gemm_wmma<false><<<dim3(1, MM / BM, 1), 128, 0, stream>>>(
      xc, x_proj_w, nullptr, dbl, MM, DTRANK + 2 * NSTATE, EDIM, EDIM,
      DTRANK + 2 * NSTATE, DTRANK + 2 * NSTATE, 0, 0, 0, 1.f, 0);

  // dt_proj: softplus(dbl[:, :64] @ [64,2048] + b) -> dlt   (A has lda=96)
  gemm_wmma<false><<<dim3(EDIM / BN, MM / BM, 1), 128, 0, stream>>>(
      dbl, dt_proj_w, dt_proj_b, dlt, MM, EDIM, DTRANK, DTRANK + 2 * NSTATE,
      EDIM, EDIM, 0, 0, 0, 1.f, 1);

  scan_kernel<<<dim3(EDIM / 256, BB), 256, 0, stream>>>(dbl, dlt, xc, xz, A_log, D_p, yv);

  // out_proj: [2048,2048] @ [2048,1024] -> m_out
  gemm_wmma<false><<<dim3(DDIM / BN, MM / BM, 1), 128, 0, stream>>>(
      yv, out_proj_w, nullptr, m_out, MM, DDIM, EDIM, EDIM, DDIM, DDIM,
      0, 0, 0, 1.f, 0);

  layernorm_kernel<<<MM, 256, 0, stream>>>(m_out, ln0_w, ln0_b, ln0o);

  // c_attn: [2048,1024] @ [1024,3072] + b -> qkv
  gemm_wmma<false><<<dim3(3 * DDIM / BN, MM / BM, 1), 128, 0, stream>>>(
      ln0o, c_attn_w, c_attn_b, qkvv, MM, 3 * DDIM, DDIM, DDIM, 3 * DDIM,
      3 * DDIM, 0, 0, 0, 1.f, 0);

  // scores = Q @ K^T / sqrt(D), per batch (B transposed: K rows over L)
  gemm_wmma<true><<<dim3(LL / BN, LL / BM, BB), 128, 0, stream>>>(
      qkvv, qkvv + DDIM, nullptr, sc, LL, LL, DDIM, 3 * DDIM, 3 * DDIM, LL,
      (long)LL * 3 * DDIM, (long)LL * 3 * DDIM, (long)LL * LL, 1.f / 32.f, 0);

  softmax_causal_kernel<<<dim3(LL, BB), 256, 0, stream>>>(sc);

  // attn @ V, per batch
  gemm_wmma<false><<<dim3(DDIM / BN, LL / BM, BB), 128, 0, stream>>>(
      sc, qkvv + 2 * DDIM, nullptr, att, LL, DDIM, LL, LL, 3 * DDIM, DDIM,
      (long)LL * LL, (long)LL * 3 * DDIM, (long)LL * DDIM, 1.f, 0);

  layernorm_kernel<<<MM, 256, 0, stream>>>(att, ln1_w, ln1_b, ln1o);

  // head: [2048,1024] @ [1024,32000] + b -> d_out (f32 logits)
  gemm_wmma<false><<<dim3(VV / BN, MM / BM, 1), 128, 0, stream>>>(
      ln1o, head_w, head_b, (float*)d_out, MM, VV, DDIM, DDIM, VV, VV,
      0, 0, 0, 1.f, 0);
}
